// TransformerBlock_34626026341067
// MI455X (gfx1250) — matrix operand order repaired and checked by reading, not compiled, not run on hardware
//
#include <hip/hip_runtime.h>
#include <hip/hip_bf16.h>
#include <math.h>

// ---------------------------------------------------------------------------
// Types for CDNA5 WMMA (wave32): bf16 16x16x32, fp32 accumulate.
// ---------------------------------------------------------------------------
typedef __attribute__((ext_vector_type(16))) __bf16 bf16x16;
typedef __attribute__((ext_vector_type(8)))  __bf16 bf16x8;
typedef __attribute__((ext_vector_type(8)))  float  f32x8;
typedef __attribute__((ext_vector_type(4)))  unsigned int v4u;
typedef __attribute__((ext_vector_type(8)))  int v8i;
typedef __attribute__((ext_vector_type(4)))  int v4i;

union FragA { bf16x16 f; bf16x8 h[2]; };

__device__ __forceinline__ f32x8 wmma_bf16(bf16x16 a, bf16x16 b, f32x8 c) {
  // (neg_a, A, neg_b, B, c_mod, C, reuse_a, reuse_b)
  return __builtin_amdgcn_wmma_f32_16x16x32_bf16(false, a, false, b, (short)0, c,
                                                 false, false);
}

// Tensor Data Mover path. This toolchain exposes the 6-arg builtin
// (probe: "too few arguments, expected 6"):
//   (uint32x4 g0, int32x8 g1, int32x4 g2, int32x4 g3, int32x8 extra, i32 cpol)
#if __has_builtin(__builtin_amdgcn_tensor_load_to_lds) && \
    __has_builtin(__builtin_amdgcn_s_wait_tensorcnt)
#define USE_TDM 1
#else
#define USE_TDM 0
#endif

#if USE_TDM
// 2D tile DMA: global (row-major, bf16, leading dim `stride_elems`) -> LDS.
// D# bit layout per CDNA5 ISA 8.3/8.4: data_size=1 (2 bytes), 2D tile.
__device__ __forceinline__ void tdm_load_2d(unsigned lds_addr, const void* gptr,
                                            unsigned tile_k, unsigned tile_rows,
                                            unsigned stride_elems,
                                            unsigned tensor_rows) {
  unsigned long long ga = (unsigned long long)(size_t)gptr;
  v4u g0;
  g0[0] = 1u;                                   // count=1 (valid user D#)
  g0[1] = lds_addr;                             // LDS byte address
  g0[2] = (unsigned)(ga & 0xffffffffu);         // global_addr[31:0]
  g0[3] = (unsigned)((ga >> 32) & 0x01ffffffu)  // global_addr[56:32]
        | 0x80000000u;                          // type=2 ("image")
  const unsigned td0 = stride_elems;            // tensor width (OOB)
  const unsigned td1 = tensor_rows;             // tensor height (OOB)
  const unsigned long long s0 = stride_elems;   // tensor_dim0_stride
  v8i g1;
  g1[0] = 0x10000;                              // data_size=1 (2B); mask/flags=0
  g1[1] = (int)((td0 & 0xffffu) << 16);         // [63:48] = tensor_dim0[15:0]
  g1[2] = (int)(((td0 >> 16) & 0xffffu) | ((td1 & 0xffffu) << 16));
  g1[3] = (int)(((td1 >> 16) & 0xffffu) | ((tile_k & 0xffffu) << 16)); // tile_dim0
  g1[4] = (int)(tile_rows & 0xffffu);           // tile_dim1 (tile_dim2=0)
  g1[5] = (int)(s0 & 0xffffffffu);              // stride0[31:0]
  g1[6] = (int)((s0 >> 32) & 0xffffu);          // stride0[47:32]
  g1[7] = 0;
  v4i z4 = {0, 0, 0, 0};
  v8i z8 = {0, 0, 0, 0, 0, 0, 0, 0};
  __builtin_amdgcn_tensor_load_to_lds(g0, g1, z4, z4, z8, 0);
}
#endif

// ---------------------------------------------------------------------------
// Transposed weight convert: out[n*K + k] = (bf16) in(k, n).
// HEADB: in is [H, K, 64] head-blocked (QKV weights), global col n = h*64+kk.
// ---------------------------------------------------------------------------
template <bool HEADB>
__global__ __launch_bounds__(256)
void cvtT_kernel(const float* __restrict__ in, __bf16* __restrict__ out,
                 int K, int N) {
  __shared__ float tile[32][33];
  const int n0 = blockIdx.x * 32, k0 = blockIdx.y * 32;
  const int c = threadIdx.x & 31, r4 = (threadIdx.x >> 5) * 4;
#pragma unroll
  for (int j = 0; j < 4; ++j) {
    int k = k0 + r4 + j, n = n0 + c;
    float v;
    if (HEADB)
      v = in[(size_t)(n >> 6) * ((size_t)K * 64) + (size_t)k * 64 + (n & 63)];
    else
      v = in[(size_t)k * N + n];
    tile[r4 + j][c] = v;
  }
  __syncthreads();
#pragma unroll
  for (int j = 0; j < 4; ++j) {
    int n = n0 + r4 + j, k = k0 + c;
    out[(size_t)n * K + k] = (__bf16)tile[c][r4 + j];
  }
}

// ---------------------------------------------------------------------------
// LayerNorm over D=1024, fp32 in -> bf16 out. One block per row.
// ---------------------------------------------------------------------------
__global__ __launch_bounds__(256)
void ln_bf16_kernel(const float* __restrict__ x, const float* __restrict__ g,
                    const float* __restrict__ be, __bf16* __restrict__ out) {
  const int D = 1024;
  const int row = blockIdx.x;
  const int tid = threadIdx.x;
  const float* xr = x + (size_t)row * D;

  float vals[4];
  float s = 0.f, s2 = 0.f;
#pragma unroll
  for (int j = 0; j < 4; ++j) {
    float v = xr[tid + 256 * j];
    vals[j] = v;
    s += v;
    s2 += v * v;
  }
  __shared__ float r1[256], r2[256];
  r1[tid] = s; r2[tid] = s2;
  __syncthreads();
  for (int o = 128; o > 0; o >>= 1) {
    if (tid < o) { r1[tid] += r1[tid + o]; r2[tid] += r2[tid + o]; }
    __syncthreads();
  }
  float mu = r1[0] * (1.0f / D);
  float var = r2[0] * (1.0f / D) - mu * mu;
  float rstd = rsqrtf(var + 1e-5f);
#pragma unroll
  for (int j = 0; j < 4; ++j) {
    int c = tid + 256 * j;
    out[(size_t)row * D + c] = (__bf16)((vals[j] - mu) * rstd * g[c] + be[c]);
  }
}

// ---------------------------------------------------------------------------
// WMMA GEMM: C = A[MxK] * Bt^T  where Bt is pre-transposed [N][K] bf16.
// Block tile 128x128, K-slab 64, double-buffered LDS; TDM fill when present.
// 8 waves, each wave 32x64 (2x4 tiles of 16x16), 16 WMMA per slab per wave.
// ---------------------------------------------------------------------------
constexpr int MODE_QK       = 0; // bf16 out, layout [B,H,S,dk]
constexpr int MODE_VT       = 1; // bf16 out, layout [B,H,dk,S] (V transposed)
constexpr int MODE_F32_RES  = 2; // f32 out = res + acc
constexpr int MODE_GELU     = 3; // bf16 out = gelu(acc + bias)  (exact erf)
constexpr int MODE_RES_BIAS = 4; // f32 out = res + acc + bias

template <int MODE>
__global__ __launch_bounds__(256)
void gemm_wmma_kernel(const __bf16* __restrict__ A, const __bf16* __restrict__ Bt,
                      const float* __restrict__ bias, const float* __restrict__ res,
                      void* __restrict__ outp, int M, int N, int K, int lda) {
  __shared__ __align__(128) __bf16 lsA[2][128][64];   // [buf][m][k]
  __shared__ __align__(128) __bf16 lsB[2][128][64];   // [buf][n][k]

  const int tid = threadIdx.x;
  const int lane = tid & 31;
  const int w  = tid >> 5;           // wave 0..7
  const int wm = w >> 1;             // 0..3
  const int wn = w & 1;              // 0..1
  const int hs = lane >> 4;          // half-wave select
  const int ln = lane & 15;

  const int n0 = blockIdx.x * 128;
  const int m0 = blockIdx.y * 128;

  const f32x8 zero = {0.f,0.f,0.f,0.f,0.f,0.f,0.f,0.f};
  f32x8 acc[2][4];
#pragma unroll
  for (int s = 0; s < 2; ++s)
#pragma unroll
    for (int t = 0; t < 4; ++t) acc[s][t] = zero;

  int buf = 0;

#if USE_TDM
  // Generic shared pointers on amdgcn carry the LDS byte offset in low 32 bits.
  const unsigned ldsA0 = (unsigned)(size_t)&lsA[0][0][0];
  const unsigned ldsB0 = (unsigned)(size_t)&lsB[0][0][0];
  const unsigned bufBytes = 128u * 64u * 2u;
  if (w == 0) {
    tdm_load_2d(ldsA0, A + (size_t)m0 * lda, 64, 128, (unsigned)lda, (unsigned)M);
    tdm_load_2d(ldsB0, Bt + (size_t)n0 * K, 64, 128, (unsigned)K, (unsigned)N);
  }
#else
  // Register-pipelined fallback: global -> regs -> LDS, double-buffered.
  const int ci0 = tid, ci1 = tid + 256;       // 512 chunks of 16 elems
  const int r0 = ci0 >> 2, c0 = (ci0 & 3) * 16;
  const int r1 = ci1 >> 2, c1 = (ci1 & 3) * 16;
  bf16x16 ra0, ra1, rb0, rb1;
  ra0 = *(const bf16x16*)(A  + (size_t)(m0 + r0) * lda + c0);
  ra1 = *(const bf16x16*)(A  + (size_t)(m0 + r1) * lda + c1);
  rb0 = *(const bf16x16*)(Bt + (size_t)(n0 + r0) * K + c0);
  rb1 = *(const bf16x16*)(Bt + (size_t)(n0 + r1) * K + c1);
#endif

  for (int k0 = 0; k0 < K; k0 += 64) {
#if USE_TDM
    if (w == 0) __builtin_amdgcn_s_wait_tensorcnt(0);
    __syncthreads();
    if (w == 0 && k0 + 64 < K) {
      unsigned nb = buf ^ 1;
      tdm_load_2d(ldsA0 + nb * bufBytes, A + (size_t)m0 * lda + (k0 + 64),
                  64, 128, (unsigned)lda, (unsigned)M);
      tdm_load_2d(ldsB0 + nb * bufBytes, Bt + (size_t)n0 * K + (k0 + 64),
                  64, 128, (unsigned)K, (unsigned)N);
    }
#else
    *(bf16x16*)&lsA[buf][r0][c0] = ra0;
    *(bf16x16*)&lsA[buf][r1][c1] = ra1;
    *(bf16x16*)&lsB[buf][r0][c0] = rb0;
    *(bf16x16*)&lsB[buf][r1][c1] = rb1;
    __syncthreads();
    if (k0 + 64 < K) {
      ra0 = *(const bf16x16*)(A  + (size_t)(m0 + r0) * lda + (k0 + 64) + c0);
      ra1 = *(const bf16x16*)(A  + (size_t)(m0 + r1) * lda + (k0 + 64) + c1);
      rb0 = *(const bf16x16*)(Bt + (size_t)(n0 + r0) * K + (k0 + 64) + c0);
      rb1 = *(const bf16x16*)(Bt + (size_t)(n0 + r1) * K + (k0 + 64) + c1);
    }
#endif
    // ---- compute current slab: 2 sub-slabs of K=32, 8 WMMA each ----
#pragma unroll
    for (int q = 0; q < 2; ++q) {
      FragA a[2];
      bf16x16 b[4];
#pragma unroll
      for (int s = 0; s < 2; ++s) {
        int r = wm * 32 + s * 16 + ln;                       // A row (lane = M)
        a[s].h[0] = *(const bf16x8*)&lsA[buf][r][q * 32 + 16 * hs];
        a[s].h[1] = *(const bf16x8*)&lsA[buf][r][q * 32 + 16 * hs + 8];
      }
#pragma unroll
      for (int t = 0; t < 4; ++t) {
        int c = wn * 64 + t * 16 + ln;                       // B col (lane = N)
        b[t] = *(const bf16x16*)&lsB[buf][c][q * 32 + 16 * hs];
      }
#pragma unroll
      for (int s = 0; s < 2; ++s)
#pragma unroll
        for (int t = 0; t < 4; ++t)
          acc[s][t] = wmma_bf16(a[s].f, b[t], acc[s][t]);
    }
    __syncthreads();
    buf ^= 1;
  }

  // ---- epilogue ----
#pragma unroll
  for (int s = 0; s < 2; ++s) {
#pragma unroll
    for (int t = 0; t < 4; ++t) {
      const int gn = n0 + wn * 64 + t * 16 + ln;
#pragma unroll
      for (int v = 0; v < 8; ++v) {
        const int gm = m0 + wm * 32 + s * 16 + hs * 8 + v;   // C row = v + 8*hs
        const float val = acc[s][t][v];
        if (MODE == MODE_QK) {
          __bf16* out = (__bf16*)outp;
          size_t idx = (size_t)((gm >> 11) * 16 + (gn >> 6)) * (2048u * 64u)
                     + (size_t)(gm & 2047) * 64 + (gn & 63);
          out[idx] = (__bf16)val;
        } else if (MODE == MODE_VT) {
          __bf16* out = (__bf16*)outp;
          size_t idx = (size_t)((gm >> 11) * 16 + (gn >> 6)) * (2048u * 64u)
                     + (size_t)(gn & 63) * 2048 + (gm & 2047);
          out[idx] = (__bf16)val;
        } else if (MODE == MODE_F32_RES) {
          float* out = (float*)outp;
          size_t idx = (size_t)gm * N + gn;
          out[idx] = res[idx] + val;
        } else if (MODE == MODE_GELU) {
          __bf16* out = (__bf16*)outp;
          float u = val + bias[gn];
          float gl = 0.5f * u * (1.0f + erff(u * 0.70710678118654752440f));
          out[(size_t)gm * N + gn] = (__bf16)gl;
        } else { // MODE_RES_BIAS
          float* out = (float*)outp;
          size_t idx = (size_t)gm * N + gn;
          out[idx] = res[idx] + val + bias[gn];
        }
      }
    }
  }
}

// ---------------------------------------------------------------------------
// Flash-style attention. Grid = B*H*(S/64) blocks, 4 waves/block.
// Each wave owns 16 query rows; online softmax; P goes through LDS to
// convert C-layout -> A-layout for the PV WMMA against V^T.
// ---------------------------------------------------------------------------
__global__ __launch_bounds__(128)
void attention_kernel(const __bf16* __restrict__ Q, const __bf16* __restrict__ Km,
                      const __bf16* __restrict__ Vt, __bf16* __restrict__ attn) {
  const int S = 2048, DK = 64;
  __shared__ __align__(64) __bf16 lsP[4][16][32];

  const int tid = threadIdx.x;
  const int lane = tid & 31;
  const int w = tid >> 5;        // wave 0..3
  const int hs = lane >> 4;
  const int ln = lane & 15;

  const int bh = blockIdx.x >> 5;    // batch*head 0..31
  const int qt = blockIdx.x & 31;    // query tile 0..31
  const int qrow = qt * 64 + w * 16;

  const __bf16* Qp = Q  + (size_t)bh * S * DK;
  const __bf16* Kp = Km + (size_t)bh * S * DK;
  const __bf16* Vp = Vt + (size_t)bh * DK * S;

  // Q fragments (16 rows x 64 k, as two 16x32 A frags), loaded once.
  FragA qa[2];
#pragma unroll
  for (int c = 0; c < 2; ++c) {
    const __bf16* qr = Qp + (size_t)(qrow + ln) * DK + c * 32;
    qa[c].h[0] = *(const bf16x8*)(qr + 16 * hs);
    qa[c].h[1] = *(const bf16x8*)(qr + 16 * hs + 8);
  }

  const f32x8 zero = {0.f,0.f,0.f,0.f,0.f,0.f,0.f,0.f};
  f32x8 o[4];
#pragma unroll
  for (int t = 0; t < 4; ++t) o[t] = zero;
  float mrow[8], lrow[8];
#pragma unroll
  for (int v = 0; v < 8; ++v) { mrow[v] = -1e30f; lrow[v] = 0.f; }

  for (int kv = 0; kv < S; kv += 32) {
    // ---- scores: S = Q K^T, two 16x16 tiles over this kv slab ----
    f32x8 sc[2];
    sc[0] = zero; sc[1] = zero;
#pragma unroll
    for (int t = 0; t < 2; ++t) {
#pragma unroll
      for (int c = 0; c < 2; ++c) {
        bf16x16 kf = *(const bf16x16*)(Kp + (size_t)(kv + t * 16 + ln) * DK
                                          + c * 32 + 16 * hs);
        sc[t] = wmma_bf16(qa[c].f, kf, sc[t]);
      }
    }
    // ---- online softmax (rows split: lanes 0-15 -> M=v, 16-31 -> M=v+8) ----
    float mnew[8], alpha[8];
#pragma unroll
    for (int v = 0; v < 8; ++v) {
      sc[0][v] *= 0.125f;            // 1/sqrt(64)
      sc[1][v] *= 0.125f;
      float m = fmaxf(sc[0][v], sc[1][v]);
      m = fmaxf(m, __shfl_xor(m, 1, 32));
      m = fmaxf(m, __shfl_xor(m, 2, 32));
      m = fmaxf(m, __shfl_xor(m, 4, 32));
      m = fmaxf(m, __shfl_xor(m, 8, 32));
      mnew[v] = fmaxf(mrow[v], m);
      alpha[v] = expf(mrow[v] - mnew[v]);
      mrow[v] = mnew[v];
    }
#pragma unroll
    for (int v = 0; v < 8; ++v) {
      float p0 = expf(sc[0][v] - mnew[v]);
      float p1 = expf(sc[1][v] - mnew[v]);
      sc[0][v] = p0; sc[1][v] = p1;
      float s = p0 + p1;
      s += __shfl_xor(s, 1, 32);
      s += __shfl_xor(s, 2, 32);
      s += __shfl_xor(s, 4, 32);
      s += __shfl_xor(s, 8, 32);
      lrow[v] = lrow[v] * alpha[v] + s;
#pragma unroll
      for (int t = 0; t < 4; ++t) o[t][v] *= alpha[v];
    }
    // ---- P: C-layout -> row-major LDS -> A fragment ----
#pragma unroll
    for (int v = 0; v < 8; ++v) {
      lsP[w][v + 8 * hs][ln]      = (__bf16)sc[0][v];
      lsP[w][v + 8 * hs][16 + ln] = (__bf16)sc[1][v];
    }
    __syncthreads();
    FragA pa;
    pa.h[0] = *(const bf16x8*)&lsP[w][ln][16 * hs];
    pa.h[1] = *(const bf16x8*)&lsP[w][ln][16 * hs + 8];
    // ---- O += P * V  (V stored transposed -> contiguous B fragments) ----
#pragma unroll
    for (int t = 0; t < 4; ++t) {
      bf16x16 vf = *(const bf16x16*)(Vp + (size_t)(t * 16 + ln) * S
                                        + kv + 16 * hs);
      o[t] = wmma_bf16(pa.f, vf, o[t]);
    }
    __syncthreads();
  }

  // ---- normalize and store concat heads: attn[b][s][h*64 + dk] ----
  const int b = bh >> 4, h = bh & 15;
#pragma unroll
  for (int v = 0; v < 8; ++v) {
    float inv = 1.0f / lrow[v];
    int srow = qrow + v + 8 * hs;
    __bf16* outr = attn + ((size_t)b * S + srow) * 1024 + h * 64;
#pragma unroll
    for (int t = 0; t < 4; ++t)
      outr[t * 16 + ln] = (__bf16)(o[t][v] * inv);
  }
}

// ---------------------------------------------------------------------------
// Host launch.
// ---------------------------------------------------------------------------
extern "C" void kernel_launch(void* const* d_in, const int* in_sizes, int n_in,
                              void* d_out, int out_size, void* d_ws, size_t ws_size,
                              hipStream_t stream) {
  (void)in_sizes; (void)n_in; (void)out_size; (void)ws_size;
  const float* x   = (const float*)d_in[0];
  const float* Wq  = (const float*)d_in[1];
  const float* Wk  = (const float*)d_in[2];
  const float* Wv  = (const float*)d_in[3];
  const float* Wo  = (const float*)d_in[4];
  const float* W1  = (const float*)d_in[5];
  const float* b1  = (const float*)d_in[6];
  const float* W2  = (const float*)d_in[7];
  const float* b2  = (const float*)d_in[8];
  const float* g1  = (const float*)d_in[9];
  const float* be1 = (const float*)d_in[10];
  const float* g2  = (const float*)d_in[11];
  const float* be2 = (const float*)d_in[12];
  float* out = (float*)d_out;

  char* ws = (char*)d_ws;
  size_t off = 0;
  auto take = [&](size_t bytes) -> char* {
    char* p = ws + off;
    off += (bytes + 255) & ~(size_t)255;
    return p;
  };

  const size_t NW = 1024u * 1024u;   // qkv/o weight elems
  const size_t NF = 1024u * 4096u;   // ffn weight elems
  const size_t NA = 4096u * 1024u;   // activation elems (B*S*D)
  const size_t NH = 4096u * 4096u;   // ffn hidden elems

  __bf16* wqT = (__bf16*)take(NW * 2);   // [N=1024][K=1024]
  __bf16* wkT = (__bf16*)take(NW * 2);
  __bf16* wvT = (__bf16*)take(NW * 2);
  __bf16* woT = (__bf16*)take(NW * 2);
  __bf16* w1T = (__bf16*)take(NF * 2);   // [N=4096][K=1024]
  __bf16* w2T = (__bf16*)take(NF * 2);   // [N=1024][K=4096]
  __bf16* h1   = (__bf16*)take(NA * 2);
  __bf16* Qb   = (__bf16*)take(NA * 2);
  __bf16* Kb   = (__bf16*)take(NA * 2);
  __bf16* Vtb  = (__bf16*)take(NA * 2);
  __bf16* attn = (__bf16*)take(NA * 2);
  float*  x1   = (float*)take(NA * 4);
  __bf16* h2   = (__bf16*)take(NA * 2);
  __bf16* f1   = (__bf16*)take(NH * 2);

  // weight convert + transpose to [N][K] bf16 (one-time per call)
  cvtT_kernel<true ><<<dim3(32, 32),  256, 0, stream>>>(Wq, wqT, 1024, 1024);
  cvtT_kernel<true ><<<dim3(32, 32),  256, 0, stream>>>(Wk, wkT, 1024, 1024);
  cvtT_kernel<true ><<<dim3(32, 32),  256, 0, stream>>>(Wv, wvT, 1024, 1024);
  cvtT_kernel<false><<<dim3(32, 32),  256, 0, stream>>>(Wo, woT, 1024, 1024);
  cvtT_kernel<false><<<dim3(128, 32), 256, 0, stream>>>(W1, w1T, 1024, 4096);
  cvtT_kernel<false><<<dim3(32, 128), 256, 0, stream>>>(W2, w2T, 4096, 1024);

  // pre-LN 1
  ln_bf16_kernel<<<4096, 256, 0, stream>>>(x, g1, be1, h1);

  // QKV projections
  gemm_wmma_kernel<MODE_QK><<<dim3(8, 32), 256, 0, stream>>>(
      h1, wqT, nullptr, nullptr, Qb, 4096, 1024, 1024, 1024);
  gemm_wmma_kernel<MODE_QK><<<dim3(8, 32), 256, 0, stream>>>(
      h1, wkT, nullptr, nullptr, Kb, 4096, 1024, 1024, 1024);
  gemm_wmma_kernel<MODE_VT><<<dim3(8, 32), 256, 0, stream>>>(
      h1, wvT, nullptr, nullptr, Vtb, 4096, 1024, 1024, 1024);

  // attention (32 bh * 32 q-tiles)
  attention_kernel<<<1024, 128, 0, stream>>>(Qb, Kb, Vtb, attn);

  // output projection + residual
  gemm_wmma_kernel<MODE_F32_RES><<<dim3(8, 32), 256, 0, stream>>>(
      attn, woT, nullptr, x, x1, 4096, 1024, 1024, 1024);

  // pre-LN 2
  ln_bf16_kernel<<<4096, 256, 0, stream>>>(x1, g2, be2, h2);

  // FFN
  gemm_wmma_kernel<MODE_GELU><<<dim3(32, 32), 256, 0, stream>>>(
      h2, w1T, b1, nullptr, f1, 4096, 4096, 1024, 1024);
  gemm_wmma_kernel<MODE_RES_BIAS><<<dim3(8, 32), 256, 0, stream>>>(
      f1, w2T, b2, x1, out, 4096, 1024, 4096, 4096);
}
